// EfficientAttention_59296318488725
// MI455X (gfx1250) — matrix operand order repaired and checked by reading, not compiled, not run on hardware
//
#include <hip/hip_runtime.h>
#include <hip/hip_bf16.h>

// ---------------------------------------------------------------------------
// EfficientAttention on MI455X (gfx1250).
//
// Big GEMMs (qkv: 16384x3072 K=1024, proj: 16384x1024 K=1024):
//   V_WMMA_F32_16X16X32_BF16 with split-precision operands
//     x = x_hi + x_lo (bf16, RN);  x@w ~= xh@wh + xh@wl + xl@wh  (f32 accum)
//   B (transposed weights) staged in LDS via GLOBAL_LOAD_ASYNC_TO_LDS_B128,
//   double-buffered on ASYNCcnt -> weight movement overlaps WMMA issue and
//   is fetched once per block instead of once per wave.
//
// Small per-head matmuls (k^T v, q*ctx; ~4 GFLOP) use full-precision
// V_WMMA_F32_16X16X4_F32 with coalesced loads.
// ---------------------------------------------------------------------------

#define B_SZ   4
#define N_SEQ  4096
#define C_DIM  1024
#define H_NUM  16
#define D_HEAD 64
#define M_TOT  (B_SZ * N_SEQ)                            // 16384 rows
#define BHND   ((size_t)B_SZ * H_NUM * N_SEQ * D_HEAD)   // 16,777,216
#define KCHUNK 32
#define NCHUNK (C_DIM / KCHUNK)                          // 32

typedef __attribute__((ext_vector_type(2)))  float  v2f;
typedef __attribute__((ext_vector_type(8)))  float  v8f;
typedef __attribute__((ext_vector_type(8)))  __bf16 v8bf;
typedef __attribute__((ext_vector_type(16))) __bf16 v16bf;

__device__ __forceinline__ v8f wmma_f32(v2f a, v2f b, v8f c) {
    return __builtin_amdgcn_wmma_f32_16x16x4_f32(
        false, a, false, b, (short)0, c, false, false);
}
__device__ __forceinline__ v8f wmma_bf16(v16bf a, v16bf b, v8f c) {
    return __builtin_amdgcn_wmma_f32_16x16x32_bf16(
        false, a, false, b, (short)0, c, false, false);
}
__device__ __forceinline__ v16bf load16bf(const __bf16* p0, const __bf16* p1) {
    v8bf a = *(const v8bf*)p0;
    v8bf b = *(const v8bf*)p1;
    return __builtin_shufflevector(a, b, 0,1,2,3,4,5,6,7,8,9,10,11,12,13,14,15);
}
__device__ __forceinline__ void split_bf(float x, __bf16& hi, __bf16& lo) {
    hi = (__bf16)x;
    lo = (__bf16)(x - (float)hi);
}
// Async copy 16B global -> LDS (GVS mode: 64-bit SGPR base + 32-bit VGPR off).
// VDST holds the LDS byte address; tracked with ASYNCcnt.
__device__ __forceinline__ void async_b128_to_lds(unsigned lds, unsigned goff,
                                                  const void* base) {
    asm volatile("global_load_async_to_lds_b128 %0, %1, %2"
                 :: "v"(lds), "v"(goff),
                    "s"((unsigned long long)(uintptr_t)base)
                 : "memory");
}
__device__ __forceinline__ void wait_async_le4() {
    asm volatile("s_wait_asynccnt 0x4" ::: "memory");
}
__device__ __forceinline__ void wait_async_le0() {
    asm volatile("s_wait_asynccnt 0x0" ::: "memory");
}

// ---------------------------------------------------------------------------
// Conversion kernels.
// ---------------------------------------------------------------------------
__global__ void __launch_bounds__(256)
split_x_kernel(const float* __restrict__ x, __bf16* __restrict__ xh,
               __bf16* __restrict__ xl)
{
    const size_t i = (size_t)blockIdx.x * 256 + threadIdx.x;
    __bf16 h, l;
    split_bf(x[i], h, l);
    xh[i] = h; xl[i] = l;
}

// W [C_DIM][NCOLS] f32  ->  Wt [NCOLS][C_DIM] bf16 hi/lo (transposed).
template <int NCOLS>
__global__ void __launch_bounds__(256)
split_wt_kernel(const float* __restrict__ w, __bf16* __restrict__ wth,
                __bf16* __restrict__ wtl)
{
    const size_t i = (size_t)blockIdx.x * 256 + threadIdx.x;   // k*NCOLS + n
    const int k = (int)(i / NCOLS), n = (int)(i % NCOLS);
    __bf16 h, l;
    split_bf(w[i], h, l);
    const size_t o = (size_t)n * C_DIM + k;
    wth[o] = h; wtl[o] = l;
}

// ---------------------------------------------------------------------------
// Split-precision bf16 GEMM with async-LDS double-buffered B tiles.
//   A row-major [M,1024] bf16 hi/lo; B transposed [NTOT][1024] bf16 hi/lo.
//   Block = 8 waves, tile 64(M) x 128(N); wave = 16 x 64.
// LDS: sB[buf][mat][col][32 k] ; col stride 64B; 4x async b128 per thread
// per chunk. Fragment layouts per CDNA5 ISA:
//   A(16x32): lane(l15,half): K = 8*half+0..7 and 16+8*half+0..7
//   B(32x16): lane(l15,half): K = 8*half+0..7 and 16+8*half+0..7 (col n = l15)
// MODE 0: permuted qkv store [3][B][H][N][D] + bias.  MODE 1: plain + bias.
// ---------------------------------------------------------------------------
template <int NTOT, int MODE>
__global__ void __launch_bounds__(256)
gemm_bf16_kernel(const __bf16* __restrict__ Ah, const __bf16* __restrict__ Al,
                 const __bf16* __restrict__ Bth, const __bf16* __restrict__ Btl,
                 const float* __restrict__ bias, float* __restrict__ out)
{
    __shared__ __align__(16) __bf16 sB[2][2][128 * KCHUNK];   // 32 KB

    const int tid  = threadIdx.x;
    const int lane = tid & 31, wid = tid >> 5;
    const int wm = wid >> 1, wn = wid & 1;
    const int m0 = blockIdx.x * 64 + wm * 16;
    const int half = lane >> 4, l15 = lane & 15;
    const int m = m0 + l15;

    const __bf16* arh = Ah + (size_t)m * C_DIM;
    const __bf16* arl = Al + (size_t)m * C_DIM;

    // --- async loader setup: 512 16B segments per matrix per chunk -------
    const __bf16* bh_base = Bth + (size_t)blockIdx.y * 128 * C_DIM;  // uniform
    const __bf16* bl_base = Btl + (size_t)blockIdx.y * 128 * C_DIM;
    const int s0 = tid, s1 = tid + 256;
    // global byte offset within the 128-col slab (col stride = 2048B)
    const unsigned go0 = (unsigned)((s0 >> 2) * 2048 + (s0 & 3) * 16);
    const unsigned go1 = (unsigned)((s1 >> 2) * 2048 + (s1 & 3) * 16);
    const unsigned sbase = (unsigned)(uintptr_t)&sB[0][0][0];
    const unsigned l00 = sbase + (unsigned)(s0 * 16);          // buf0, hi
    const unsigned l01 = sbase + (unsigned)(s1 * 16);

    auto issue = [&](int chunk, int buf) {
        const unsigned koff = (unsigned)(chunk * KCHUNK * 2);  // bytes along k
        const unsigned lb = (unsigned)(buf * 16384);
        async_b128_to_lds(l00 + lb,        go0 + koff, bh_base);
        async_b128_to_lds(l01 + lb,        go1 + koff, bh_base);
        async_b128_to_lds(l00 + lb + 8192, go0 + koff, bl_base);
        async_b128_to_lds(l01 + lb + 8192, go1 + koff, bl_base);
    };

    v8f acc[4] = {};
    issue(0, 0);
    for (int c = 0; c < NCHUNK; ++c) {
        const int buf = c & 1;
        if (c + 1 < NCHUNK) {
            issue(c + 1, buf ^ 1);
            wait_async_le4();      // in-order: oldest 4 (chunk c) have landed
        } else {
            wait_async_le0();
        }
        __syncthreads();

        const int k0 = c * KCHUNK;
        const int ka = k0 + 8 * half;
        // prefetch next A chunk into cache while this chunk computes
        if (c + 1 < NCHUNK) {
            __builtin_prefetch(arh + ka + KCHUNK, 0, 0);
            __builtin_prefetch(arl + ka + KCHUNK, 0, 0);
        }
        const v16bf a_hi = load16bf(arh + ka, arh + ka + 16);
        const v16bf a_lo = load16bf(arl + ka, arl + ka + 16);
#pragma unroll
        for (int t = 0; t < 4; ++t) {
            const int col = wn * 64 + t * 16 + l15;            // local column
            const __bf16* ph = &sB[buf][0][col * KCHUNK + 8 * half];
            const __bf16* pl = &sB[buf][1][col * KCHUNK + 8 * half];
            const v16bf b_hi = load16bf(ph, ph + 16);
            const v16bf b_lo = load16bf(pl, pl + 16);
            acc[t] = wmma_bf16(a_hi, b_hi, acc[t]);
            acc[t] = wmma_bf16(a_hi, b_lo, acc[t]);
            acc[t] = wmma_bf16(a_lo, b_hi, acc[t]);
        }
        __syncthreads();   // protect buf from being overwritten at c+2
    }

    const int n0 = blockIdx.y * 128 + wn * 64;
#pragma unroll
    for (int t = 0; t < 4; ++t) {
        const int col = n0 + t * 16 + l15;
        const float bb = bias[col];
        if (MODE == 0) {   // scatter into [s][b][h][n][d]
            const int s = col >> 10, cc = col & 1023;
            const int h = cc >> 6,   d = cc & 63;
#pragma unroll
            for (int r = 0; r < 8; ++r) {
                const int row = m0 + r + 8 * half;
                const int b = row >> 12, n = row & 4095;
                const size_t idx =
                    ((((size_t)s * B_SZ + b) * H_NUM + h) * N_SEQ + n) * D_HEAD + d;
                out[idx] = acc[t][r] + bb;
            }
        } else {           // plain row-major store
#pragma unroll
            for (int r = 0; r < 8; ++r) {
                const int row = m0 + r + 8 * half;
                out[(size_t)row * NTOT + col] = acc[t][r] + bb;
            }
        }
    }
}

// ---------------------------------------------------------------------------
// softmax(q) over head_dim (rows of 64). One wave per row.
// ---------------------------------------------------------------------------
__global__ void __launch_bounds__(256)
q_softmax_kernel(float* __restrict__ q)
{
    const int lane = threadIdx.x & 31;
    const int wid  = threadIdx.x >> 5;
    const size_t row = (size_t)blockIdx.x * 8 + wid;
    float* p = q + row * D_HEAD;

    float x0 = p[lane], x1 = p[lane + 32];
    float mx = fmaxf(x0, x1);
#pragma unroll
    for (int off = 16; off; off >>= 1) mx = fmaxf(mx, __shfl_xor(mx, off, 32));
    const float e0 = __expf(x0 - mx), e1 = __expf(x1 - mx);
    float s = e0 + e1;
#pragma unroll
    for (int off = 16; off; off >>= 1) s += __shfl_xor(s, off, 32);
    const float inv = 1.0f / s;
    p[lane] = e0 * inv;
    p[lane + 32] = e1 * inv;
}

// ---------------------------------------------------------------------------
// softmax(k) over sequence N (per (b,h,d) column). One block per (b,h).
// ---------------------------------------------------------------------------
__global__ void __launch_bounds__(256)
k_softmax_kernel(float* __restrict__ k)
{
    float* base = k + (size_t)blockIdx.x * N_SEQ * D_HEAD;
    const int d = threadIdx.x & 63;
    const int g = threadIdx.x >> 6;
    __shared__ float red[4][64];

    float mx = -INFINITY;
    for (int n = g; n < N_SEQ; n += 4) mx = fmaxf(mx, base[n * D_HEAD + d]);
    red[g][d] = mx;
    __syncthreads();
    const float cmx = fmaxf(fmaxf(red[0][d], red[1][d]),
                            fmaxf(red[2][d], red[3][d]));
    __syncthreads();

    float sum = 0.f;
    for (int n = g; n < N_SEQ; n += 4) sum += __expf(base[n * D_HEAD + d] - cmx);
    red[g][d] = sum;
    __syncthreads();
    const float inv = 1.0f / (red[0][d] + red[1][d] + red[2][d] + red[3][d]);

    for (int n = g; n < N_SEQ; n += 4) {
        const float vv = base[n * D_HEAD + d];
        base[n * D_HEAD + d] = __expf(vv - cmx) * inv;
    }
}

// ---------------------------------------------------------------------------
// context[b,h] = k_sm^T @ v  (64x64, K=4096), full fp32 WMMA, coalesced.
// ---------------------------------------------------------------------------
__global__ void __launch_bounds__(512)
context_kernel(const float* __restrict__ k, const float* __restrict__ v,
               float* __restrict__ ctx)
{
    const int lane = threadIdx.x & 31, wid = threadIdx.x >> 5;
    const int tm = wid >> 2, tn = wid & 3;
    const size_t off = (size_t)blockIdx.x * N_SEQ * D_HEAD;
    const float* kb = k + off;
    const float* vb = v + off;
    const int half = lane >> 4, l15 = lane & 15;
    const int dm = tm * 16 + l15;
    const int en = tn * 16 + l15;

    v8f acc = {};
    for (int n0 = 0; n0 < N_SEQ; n0 += 4) {
        const int kk = n0 + 2 * half;
        v2f a, b;
        a.x = kb[(size_t)kk * D_HEAD + dm];
        a.y = kb[(size_t)(kk + 1) * D_HEAD + dm];
        b.x = vb[(size_t)kk * D_HEAD + en];
        b.y = vb[(size_t)(kk + 1) * D_HEAD + en];
        acc = wmma_f32(a, b, acc);
    }

    float* cb = ctx + (size_t)blockIdx.x * D_HEAD * D_HEAD;
#pragma unroll
    for (int r = 0; r < 8; ++r) {
        const int row = tm * 16 + r + 8 * half;
        cb[row * D_HEAD + en] = acc[r];
    }
}

// ---------------------------------------------------------------------------
// attn = q_sm @ context (f32 WMMA, K=64); epilogue writes bf16 hi/lo split
// directly in [B][N][C] layout so the proj GEMM needs no permutation.
// ---------------------------------------------------------------------------
__global__ void __launch_bounds__(256)
attn_kernel(const float* __restrict__ q, const float* __restrict__ ctx,
            __bf16* __restrict__ ohi, __bf16* __restrict__ olo)
{
    const int lane = threadIdx.x & 31, wid = threadIdx.x >> 5;
    const int bh = blockIdx.y;                 // 0..63
    const int rb = blockIdx.x;                 // 0..31
    const int half = lane >> 4, l15 = lane & 15;
    const int b = bh >> 4, h = bh & 15;

    const float* qb = q + (size_t)bh * N_SEQ * D_HEAD;
    const float* cb = ctx + (size_t)bh * D_HEAD * D_HEAD;
    const int row_base = rb * 128 + wid * 16;
    const float* qrow = qb + (size_t)(row_base + l15) * D_HEAD;

    v8f acc[4] = {};
    for (int k0 = 0; k0 < D_HEAD; k0 += 4) {
        const int kk = k0 + 2 * half;
        const v2f a = *(const v2f*)(qrow + kk);
#pragma unroll
        for (int t = 0; t < 4; ++t) {
            const int e = t * 16 + l15;
            v2f bf;
            bf.x = cb[kk * D_HEAD + e];
            bf.y = cb[(kk + 1) * D_HEAD + e];
            acc[t] = wmma_f32(a, bf, acc[t]);
        }
    }
#pragma unroll
    for (int t = 0; t < 4; ++t)
#pragma unroll
        for (int r = 0; r < 8; ++r) {
            const int n = row_base + r + 8 * half;
            const int e = t * 16 + l15;
            const size_t idx = ((size_t)b * N_SEQ + n) * C_DIM + h * D_HEAD + e;
            __bf16 hi, lo;
            split_bf(acc[t][r], hi, lo);
            ohi[idx] = hi; olo[idx] = lo;
        }
}

// ---------------------------------------------------------------------------
// Host launch. Workspace: q,k,v f32 (64MB ea) | ctx (1MB) | x hi/lo bf16
// (32MB ea) | Wqkv_t hi/lo (6MB ea) | Wproj_t hi/lo (2MB ea) | attn hi/lo
// bf16 (32MB ea)  => ~337MB total from d_ws.
// ---------------------------------------------------------------------------
extern "C" void kernel_launch(void* const* d_in, const int* in_sizes, int n_in,
                              void* d_out, int out_size, void* d_ws, size_t ws_size,
                              hipStream_t stream)
{
    (void)in_sizes; (void)n_in; (void)out_size; (void)ws_size;

    const float* x      = (const float*)d_in[0];
    const float* W_qkv  = (const float*)d_in[1];
    const float* b_qkv  = (const float*)d_in[2];
    const float* W_proj = (const float*)d_in[3];
    const float* b_proj = (const float*)d_in[4];
    float* out = (float*)d_out;

    float* qbuf = (float*)d_ws;                       // BHND
    float* kbuf = qbuf + BHND;
    float* vbuf = kbuf + BHND;
    float* ctx  = vbuf + BHND;                        // B*H*D*D
    __bf16* p   = (__bf16*)(ctx + (size_t)B_SZ * H_NUM * D_HEAD * D_HEAD);
    __bf16* xh   = p; p += (size_t)M_TOT * C_DIM;
    __bf16* xl   = p; p += (size_t)M_TOT * C_DIM;
    __bf16* wqh  = p; p += (size_t)(3 * C_DIM) * C_DIM;
    __bf16* wql  = p; p += (size_t)(3 * C_DIM) * C_DIM;
    __bf16* wph  = p; p += (size_t)C_DIM * C_DIM;
    __bf16* wpl  = p; p += (size_t)C_DIM * C_DIM;
    __bf16* ath  = p; p += (size_t)M_TOT * C_DIM;
    __bf16* atl  = p;

    // 0) precision split (+ weight transpose)
    split_x_kernel<<<dim3((unsigned)((size_t)M_TOT * C_DIM / 256)), 256, 0, stream>>>(
        x, xh, xl);
    split_wt_kernel<3 * C_DIM><<<dim3((unsigned)((size_t)C_DIM * 3 * C_DIM / 256)),
                                 256, 0, stream>>>(W_qkv, wqh, wql);
    split_wt_kernel<C_DIM><<<dim3((unsigned)((size_t)C_DIM * C_DIM / 256)),
                             256, 0, stream>>>(W_proj, wph, wpl);
    // 1) qkv GEMM (bf16 split WMMA + async-LDS B tiles), permuted store + bias
    gemm_bf16_kernel<3 * C_DIM, 0><<<dim3(M_TOT / 64, (3 * C_DIM) / 128), 256, 0,
                                     stream>>>(xh, xl, wqh, wql, b_qkv, qbuf);
    // 2) softmax(q) over D
    q_softmax_kernel<<<dim3((unsigned)(BHND / D_HEAD / 8)), 256, 0, stream>>>(qbuf);
    // 3) softmax(k) over N
    k_softmax_kernel<<<dim3(B_SZ * H_NUM), 256, 0, stream>>>(kbuf);
    // 4) context = k_sm^T @ v (f32 WMMA)
    context_kernel<<<dim3(B_SZ * H_NUM), 512, 0, stream>>>(kbuf, vbuf, ctx);
    // 5) attn = q_sm @ context (f32 WMMA), bf16 hi/lo store in [B,N,C]
    attn_kernel<<<dim3(N_SEQ / 128, B_SZ * H_NUM), 256, 0, stream>>>(
        qbuf, ctx, ath, atl);
    // 6) final = attn @ W_proj + bias (bf16 split WMMA) -> d_out
    gemm_bf16_kernel<C_DIM, 1><<<dim3(M_TOT / 64, C_DIM / 128), 256, 0, stream>>>(
        ath, atl, wph, wpl, b_proj, out);
}
